// RWKV7TimeMixing_59133109731576
// MI455X (gfx1250) — matrix operand order repaired and checked by reading, not compiled, not run on hardware
//
#include <hip/hip_runtime.h>
#include <hip/hip_bf16.h>

#define BB 8
#define TT 1024
#define CC 768
#define HH 12
#define NN 64
#define BT (BB*TT)                  /* 8192 tokens */
#define BTC ((size_t)BT*CC)         /* 6291456 */
#define CORE_N 153
#define CORE_P 160
#define GHOST_N 615
#define GHOST_P 624

typedef __attribute__((ext_vector_type(16))) _Float16 v16h;
typedef __attribute__((ext_vector_type(8)))  float    v8f;

union HV { v16h v; unsigned int u[8]; };

// ---------------- weight convert + pad to f16 (row-major copy) ----------------
__global__ __launch_bounds__(256) void cvt_pad_kernel(const float* __restrict__ src,
    _Float16* __restrict__ dst, int rows, int cols, int cpad, long total)
{
  long i = (long)blockIdx.x*256 + threadIdx.x;
  if (i >= total) return;
  int r = (int)(i / cpad), c = (int)(i % cpad);
  float v = (r < rows && c < cols) ? src[(long)r*cols + c] : 0.f;
  dst[i] = (_Float16)v;
}

// -------- weight convert with transpose: src[R,C] f32 -> dst[C,R] f16 --------
__global__ __launch_bounds__(256) void cvt_t_kernel(const float* __restrict__ src,
    _Float16* __restrict__ dst, int R, int C, long total)
{
  long i = (long)blockIdx.x*256 + threadIdx.x;   // index into dst [C,R]
  if (i >= total) return;
  int c = (int)(i / R), r = (int)(i % R);
  dst[i] = (_Float16)src[(long)r*C + c];
}

// ---------------- token mixing: xr..xg in f16 ----------------
__global__ __launch_bounds__(256) void mix6_kernel(const float* __restrict__ x,
    const float* __restrict__ xp,
    const float* __restrict__ mr, const float* __restrict__ mw, const float* __restrict__ mk,
    const float* __restrict__ mv, const float* __restrict__ ma, const float* __restrict__ mg,
    _Float16* __restrict__ xr, _Float16* __restrict__ xw, _Float16* __restrict__ xk,
    _Float16* __restrict__ xv, _Float16* __restrict__ xa, _Float16* __restrict__ xg)
{
  size_t i = (size_t)blockIdx.x*256 + threadIdx.x;
  if (i >= BTC) return;
  int c = (int)(i % CC);
  float xi = x[i], d = xp[i] - xi;
  xr[i] = (_Float16)(xi + d*mr[c]);
  xw[i] = (_Float16)(xi + d*mw[c]);
  xk[i] = (_Float16)(xi + d*mk[c]);
  xv[i] = (_Float16)(xi + d*mv[c]);
  xa[i] = (_Float16)(xi + d*ma[c]);
  xg[i] = (_Float16)(xi + d*mg[c]);
}

// ---------------- WMMA GEMM: one wave computes 4 N-tiles of one M-tile --------
__global__ __launch_bounds__(32) void gemm_wmma_kernel(
    const _Float16* __restrict__ A, int lda,
    const _Float16* __restrict__ Bm, int ldb, int bRows, int K,
    float* __restrict__ outF, int ooff, int nactF, const float* __restrict__ bias,
    _Float16* __restrict__ outH, int ldh, int nactH, int actH)
{
  const int lane = threadIdx.x & 31;
  const int sub  = lane & 15;
  const int hi   = lane >> 4;
  const int m0    = blockIdx.x * 16;
  const int nbase = blockIdx.y * 64;

  const _Float16* Arow = A + (size_t)(m0 + sub)*lda + (hi << 3);
  const _Float16* Brow[4];
#pragma unroll
  for (int u = 0; u < 4; ++u) {
    int r = nbase + u*16 + sub;
    r = (r < bRows) ? r : (bRows - 1);
    Brow[u] = Bm + (size_t)r*ldb + (hi << 3);
  }

  v8f acc[4] = {};
  for (int k0 = 0; k0 < K; k0 += 32) {
    HV a;
#pragma unroll
    for (int q = 0; q < 8; ++q)
      a.u[q] = *(const unsigned int*)(Arow + k0 + ((q & 4) << 2) + ((q & 3) << 1));
#pragma unroll
    for (int u = 0; u < 4; ++u) {
      HV b;
#pragma unroll
      for (int q = 0; q < 8; ++q)
        b.u[q] = *(const unsigned int*)(Brow[u] + k0 + ((q & 4) << 2) + ((q & 3) << 1));
      acc[u] = __builtin_amdgcn_wmma_f32_16x16x32_f16(false, a.v, false, b.v,
                                                      (short)0, acc[u], false, false);
    }
  }

#pragma unroll
  for (int u = 0; u < 4; ++u) {
    const int col = nbase + u*16 + sub;
    if (outF != nullptr && col < nactF) {
      float bb = bias ? bias[ooff + col] : 0.f;
#pragma unroll
      for (int q = 0; q < 8; ++q)
        outF[(size_t)(m0 + q + (hi << 3))*CC + ooff + col] = acc[u][q] + bb;
    }
    if (outH != nullptr && col < ldh) {
#pragma unroll
      for (int q = 0; q < 8; ++q) {
        float v = (col < nactH) ? acc[u][q] : 0.f;
        if (actH == 1) v = tanhf(v);
        else if (actH == 2) v = 1.f/(1.f + __expf(-v));
        outH[(size_t)(m0 + q + (hi << 3))*ldh + col] = (_Float16)v;
      }
    }
  }
}

// ---------------- per-(token,head) nonlinearities + kk norm ----------------
__global__ __launch_bounds__(64) void pass2_kernel(
    float* __restrict__ kbuf, float* __restrict__ vbuf,
    float* __restrict__ abuf, float* __restrict__ wbuf,
    const float* __restrict__ vpre, float* __restrict__ kkbuf,
    const float* __restrict__ a0, const float* __restrict__ w0, const float* __restrict__ v0,
    const float* __restrict__ k_k, const float* __restrict__ k_a,
    const float* __restrict__ v_first)
{
  int blk = blockIdx.x;            // tok*H + h
  int h = blk % HH;
  size_t tok = (size_t)(blk / HH);
  int n = threadIdx.x;
  int c = h*NN + n;
  size_t idx = tok*CC + c;
  float k = kbuf[idx];
  float a = 1.f/(1.f + __expf(-(a0[c] + abuf[idx])));
  float w = __expf(-0.606531f / (1.f + __expf(-(w0[c] + wbuf[idx]))));
  float kkv = k * k_k[c];
  __shared__ float red[NN];
  red[n] = kkv*kkv;
  __syncthreads();
  for (int s = 32; s > 0; s >>= 1) { if (n < s) red[n] += red[n+s]; __syncthreads(); }
  float nrm = fmaxf(sqrtf(red[0]), 1e-12f);
  float sv = 1.f/(1.f + __expf(-(v0[c] + vpre[idx])));
  float vv = vbuf[idx];
  vv = vv + (v_first[idx] - vv)*sv;
  kbuf[idx]  = k * (1.f + (a - 1.f)*k_a[c]);
  vbuf[idx]  = vv;
  abuf[idx]  = a;
  wbuf[idx]  = w;
  kkbuf[idx] = kkv / nrm;
}

// -------- async global->LDS load (CDNA5, tracked by ASYNCcnt) --------
__device__ __forceinline__ void async_ld_f32(unsigned lds_off, const float* g) {
  asm volatile("global_load_async_to_lds_b32 %0, %1, off"
               :: "v"(lds_off), "v"(g) : "memory");
}

// ---------------- recurrent scan: one block per (b,h), thread = row i ----------
// Double-buffered: async-DMA step t+1's six 64-float vectors into LDS while the
// VALU computes step t; s_wait_asynccnt 6 retires only the oldest batch.
__global__ __launch_bounds__(64) void scan_kernel(
    const float* __restrict__ rbuf, const float* __restrict__ kbuf,
    const float* __restrict__ vbuf, const float* __restrict__ kkbuf,
    const float* __restrict__ abuf, const float* __restrict__ wbuf,
    const float* __restrict__ state_in, float* __restrict__ ybuf,
    float* __restrict__ state_out)
{
  int bh = blockIdx.x;
  int b = bh / HH, h = bh % HH;
  int i = threadIdx.x;
  float st[NN];
  const float* s0 = state_in + (size_t)bh*NN*NN + (size_t)i*NN;
#pragma unroll
  for (int j = 0; j < NN; ++j) st[j] = s0[j];

  __shared__ float buf[2][6][NN];
  size_t base = (size_t)b*TT*CC + (size_t)h*NN;
  const float* srcs[6] = {rbuf, kbuf, vbuf, kkbuf, abuf, wbuf};

  auto issue = [&](int tt, int nb){
    size_t off = base + (size_t)tt*CC + i;
#pragma unroll
    for (int v = 0; v < 6; ++v)
      async_ld_f32((unsigned)(uintptr_t)&buf[nb][v][i], srcs[v] + off);
  };

  issue(0, 0);
  for (int t = 0; t < TT; ++t) {
    int pb = t & 1;
    // prefetch next step's vectors (final iteration over-reads one row into the
    // adjacent workspace buffer; that batch is never consumed)
    issue(t + 1, pb ^ 1);
    asm volatile("s_wait_asynccnt 6" ::: "memory");   // oldest batch (step t) done
    __syncthreads();
    const float* sr  = buf[pb][0];
    const float* sk  = buf[pb][1];
    const float* sv  = buf[pb][2];
    const float* skk = buf[pb][3];
    const float* sa  = buf[pb][4];
    const float* sw  = buf[pb][5];
    float sav = 0.f;
#pragma unroll
    for (int j = 0; j < NN; ++j) sav -= st[j]*skk[j];
    float vi = sv[i], wi = sw[i], y = 0.f;
#pragma unroll
    for (int j = 0; j < NN; ++j) {
      float s = st[j]*wi + sav*(skk[j]*sa[j]) + vi*sk[j];
      st[j] = s;
      y += s*sr[j];
    }
    ybuf[base + (size_t)t*CC + i] = y;
    __syncthreads();
  }
  float* so = state_out + (size_t)bh*NN*NN + (size_t)i*NN;
#pragma unroll
  for (int j = 0; j < NN; ++j) so[j] = st[j];
}

// ---------------- groupnorm + bonus + gate ----------------
__global__ __launch_bounds__(64) void post_kernel(
    const float* __restrict__ ybuf, const float* __restrict__ rbuf,
    const float* __restrict__ kbuf, const float* __restrict__ vbuf,
    const float* __restrict__ gbuf,
    const float* __restrict__ ln_w, const float* __restrict__ ln_b,
    const float* __restrict__ r_k, _Float16* __restrict__ gout)
{
  int blk = blockIdx.x;
  int h = blk % HH;
  size_t tok = (size_t)(blk / HH);
  int n = threadIdx.x;
  int c = h*NN + n;
  size_t idx = tok*CC + c;
  float y = ybuf[idx];
  __shared__ float red[NN];
  red[n] = y; __syncthreads();
  for (int s = 32; s > 0; s >>= 1) { if (n < s) red[n] += red[n+s]; __syncthreads(); }
  float mu = red[0] * (1.f/NN);
  __syncthreads();
  float d = y - mu;
  red[n] = d*d; __syncthreads();
  for (int s = 32; s > 0; s >>= 1) { if (n < s) red[n] += red[n+s]; __syncthreads(); }
  float var = red[0] * (1.f/NN);
  float outv = d * rsqrtf(var + 0.00064f) * ln_w[c] + ln_b[c];
  __syncthreads();
  red[n] = rbuf[idx]*kbuf[idx]*r_k[c];   // r_k is [H,N] -> flat index h*64+n == c
  __syncthreads();
  for (int s = 32; s > 0; s >>= 1) { if (n < s) red[n] += red[n+s]; __syncthreads(); }
  outv += red[0] * vbuf[idx];
  gout[idx] = (_Float16)(outv * gbuf[idx]);
}

// ---------------- tail copies ----------------
__global__ __launch_bounds__(256) void copyf_kernel(const float* __restrict__ src,
    float* __restrict__ dst, size_t nElem)
{
  size_t i = (size_t)blockIdx.x*256 + threadIdx.x;
  if (i < nElem) dst[i] = src[i];
}

__global__ __launch_bounds__(256) void xlast_kernel(const float* __restrict__ x,
    float* __restrict__ dst)
{
  int i = blockIdx.x*256 + threadIdx.x;
  if (i >= BB*CC) return;
  int b = i / CC, c = i % CC;
  dst[i] = x[((size_t)b*TT + (TT-1))*CC + c];
}

extern "C" void kernel_launch(void* const* d_in, const int* in_sizes, int n_in,
                              void* d_out, int out_size, void* d_ws, size_t ws_size,
                              hipStream_t stream) {
  (void)in_sizes; (void)n_in; (void)out_size; (void)ws_size;
  const float* x      = (const float*)d_in[0];
  const float* x_prev = (const float*)d_in[1];
  const float* state  = (const float*)d_in[2];
  const float* vfirst = (const float*)d_in[3];
  const float* x_r = (const float*)d_in[4];
  const float* x_w = (const float*)d_in[5];
  const float* x_k = (const float*)d_in[6];
  const float* x_v = (const float*)d_in[7];
  const float* x_a = (const float*)d_in[8];
  const float* x_g = (const float*)d_in[9];
  const float* w0 = (const float*)d_in[10];
  const float* w1 = (const float*)d_in[11];
  const float* w2 = (const float*)d_in[12];
  const float* a0 = (const float*)d_in[13];
  const float* a1 = (const float*)d_in[14];
  const float* a2 = (const float*)d_in[15];
  const float* v0 = (const float*)d_in[16];
  const float* v1 = (const float*)d_in[17];
  const float* v2 = (const float*)d_in[18];
  const float* g1 = (const float*)d_in[19];
  const float* g2 = (const float*)d_in[20];
  const float* k_k = (const float*)d_in[21];
  const float* k_a = (const float*)d_in[22];
  const float* r_k = (const float*)d_in[23];
  const float* rec_w = (const float*)d_in[24];
  const float* rec_g = (const float*)d_in[25];
  const float* rec_b = (const float*)d_in[26];
  const float* key_w = (const float*)d_in[27];
  const float* key_g = (const float*)d_in[28];
  const float* key_b = (const float*)d_in[29];
  const float* val_w = (const float*)d_in[30];
  const float* val_g = (const float*)d_in[31];
  const float* val_b = (const float*)d_in[32];
  const float* out_w = (const float*)d_in[33];
  const float* out_g = (const float*)d_in[34];
  const float* out_b = (const float*)d_in[35];
  const float* ln_w = (const float*)d_in[36];
  const float* ln_b = (const float*)d_in[37];

  float* out = (float*)d_out;
  const size_t OFF_XLAST  = BTC;
  const size_t OFF_STATE  = OFF_XLAST + (size_t)BB*CC;
  const size_t OFF_VFIRST = OFF_STATE + (size_t)BB*HH*NN*NN;

  // ---- workspace bump allocator ----
  char* wp = (char*)d_ws;
  auto alloc = [&](size_t bytes) -> void* {
    void* q = (void*)wp;
    wp += (bytes + 255) & ~(size_t)255;
    return q;
  };
  _Float16* xr_h = (_Float16*)alloc(BTC*2);
  _Float16* xw_h = (_Float16*)alloc(BTC*2);
  _Float16* xk_h = (_Float16*)alloc(BTC*2);
  _Float16* xv_h = (_Float16*)alloc(BTC*2);
  _Float16* xa_h = (_Float16*)alloc(BTC*2);
  _Float16* xg_h = (_Float16*)alloc(BTC*2);
  _Float16* core_h = (_Float16*)alloc((size_t)BT*CORE_P*2);
  _Float16* t_h    = (_Float16*)alloc((size_t)BT*128*2);
  _Float16* gout_h = (_Float16*)alloc(BTC*2);
  _Float16* recw_h = (_Float16*)alloc((size_t)CORE_P*CC*2);
  _Float16* keyw_h = (_Float16*)alloc((size_t)CORE_P*CC*2);
  _Float16* valw_h = (_Float16*)alloc((size_t)CORE_P*CC*2);
  _Float16* outw_h = (_Float16*)alloc((size_t)CORE_P*CC*2);
  _Float16* recg_h = (_Float16*)alloc((size_t)GHOST_P*CORE_P*2);
  _Float16* keyg_h = (_Float16*)alloc((size_t)GHOST_P*CORE_P*2);
  _Float16* valg_h = (_Float16*)alloc((size_t)GHOST_P*CORE_P*2);
  _Float16* outg_h = (_Float16*)alloc((size_t)GHOST_P*CORE_P*2);
  _Float16* w1t_h = (_Float16*)alloc((size_t)32*CC*2);    // [32,768]
  _Float16* w2t_h = (_Float16*)alloc((size_t)CC*32*2);    // [768,32]
  _Float16* a1t_h = (_Float16*)alloc((size_t)32*CC*2);
  _Float16* a2t_h = (_Float16*)alloc((size_t)CC*32*2);
  _Float16* v1t_h = (_Float16*)alloc((size_t)32*CC*2);
  _Float16* v2t_h = (_Float16*)alloc((size_t)CC*32*2);
  _Float16* g1t_h = (_Float16*)alloc((size_t)128*CC*2);   // [128,768]
  _Float16* g2t_h = (_Float16*)alloc((size_t)CC*128*2);   // [768,128]
  float* rbuf  = (float*)alloc(BTC*4);
  float* kbuf  = (float*)alloc(BTC*4);
  float* vbuf  = (float*)alloc(BTC*4);
  float* wbuf  = (float*)alloc(BTC*4);
  float* abuf  = (float*)alloc(BTC*4);
  float* vpre  = (float*)alloc(BTC*4);
  float* gbuf  = (float*)alloc(BTC*4);
  float* kkbuf = (float*)alloc(BTC*4);
  float* ybuf  = (float*)alloc(BTC*4);

  auto cvt = [&](const float* s, _Float16* dd, int rows, int cols, int rpad, int cpad){
    long tot = (long)rpad*cpad;
    cvt_pad_kernel<<<dim3((unsigned)((tot+255)/256)), 256, 0, stream>>>(s, dd, rows, cols, cpad, tot);
  };
  auto cvtT = [&](const float* s, _Float16* dd, int R, int C){
    long tot = (long)R*C;
    cvt_t_kernel<<<dim3((unsigned)((tot+255)/256)), 256, 0, stream>>>(s, dd, R, C, tot);
  };
  cvt(rec_w, recw_h, CORE_N, CC, CORE_P, CC);
  cvt(key_w, keyw_h, CORE_N, CC, CORE_P, CC);
  cvt(val_w, valw_h, CORE_N, CC, CORE_P, CC);
  cvt(out_w, outw_h, CORE_N, CC, CORE_P, CC);
  cvt(rec_g, recg_h, GHOST_N, CORE_N, GHOST_P, CORE_P);
  cvt(key_g, keyg_h, GHOST_N, CORE_N, GHOST_P, CORE_P);
  cvt(val_g, valg_h, GHOST_N, CORE_N, GHOST_P, CORE_P);
  cvt(out_g, outg_h, GHOST_N, CORE_N, GHOST_P, CORE_P);
  cvtT(w1, w1t_h, CC, 32);    // [768,32] -> [32,768]
  cvtT(w2, w2t_h, 32, CC);    // [32,768] -> [768,32]
  cvtT(a1, a1t_h, CC, 32);
  cvtT(a2, a2t_h, 32, CC);
  cvtT(v1, v1t_h, CC, 32);
  cvtT(v2, v2t_h, 32, CC);
  cvtT(g1, g1t_h, CC, 128);   // -> [128,768]
  cvtT(g2, g2t_h, 128, CC);   // -> [768,128]

  mix6_kernel<<<dim3((unsigned)((BTC+255)/256)), 256, 0, stream>>>(
      x, x_prev, x_r, x_w, x_k, x_v, x_a, x_g,
      xr_h, xw_h, xk_h, xv_h, xa_h, xg_h);

  const int MT = BT/16;   // 512 M-tiles
  auto gemm = [&](const _Float16* A, int lda, const _Float16* Bm, int ldb, int bRows, int K,
                  int ntiles, float* oF, int ooff, int nF, const float* bias,
                  _Float16* oH, int ldh, int nH, int actH){
    gemm_wmma_kernel<<<dim3(MT, (ntiles + 3)/4), 32, 0, stream>>>(
        A, lda, Bm, ldb, bRows, K, oF, ooff, nF, bias, oH, ldh, nH, actH);
  };

  // Ghost linears: r, k, v  (stage1: 768->153 + f16 core; stage2: 153(pad160)->615)
  gemm(xr_h, CC, recw_h, CC, CORE_P, CC, CORE_P/16, rbuf, 0, CORE_N, rec_b, core_h, CORE_P, CORE_N, 0);
  gemm(core_h, CORE_P, recg_h, CORE_P, GHOST_P, CORE_P, GHOST_P/16, rbuf, CORE_N, GHOST_N, rec_b, nullptr, 0, 0, 0);
  gemm(xk_h, CC, keyw_h, CC, CORE_P, CC, CORE_P/16, kbuf, 0, CORE_N, key_b, core_h, CORE_P, CORE_N, 0);
  gemm(core_h, CORE_P, keyg_h, CORE_P, GHOST_P, CORE_P, GHOST_P/16, kbuf, CORE_N, GHOST_N, key_b, nullptr, 0, 0, 0);
  gemm(xv_h, CC, valw_h, CC, CORE_P, CC, CORE_P/16, vbuf, 0, CORE_N, val_b, core_h, CORE_P, CORE_N, 0);
  gemm(core_h, CORE_P, valg_h, CORE_P, GHOST_P, CORE_P, GHOST_P/16, vbuf, CORE_N, GHOST_N, val_b, nullptr, 0, 0, 0);

  // loras (all weights pre-transposed to [N,K])
  gemm(xw_h, CC, w1t_h, CC, 32, CC, 2,  nullptr, 0, 0, nullptr, t_h, 32, 32, 1);     // tanh
  gemm(t_h, 32, w2t_h, 32, CC, 32, CC/16, wbuf, 0, CC, nullptr, nullptr, 0, 0, 0);
  gemm(xa_h, CC, a1t_h, CC, 32, CC, 2,  nullptr, 0, 0, nullptr, t_h, 32, 32, 0);
  gemm(t_h, 32, a2t_h, 32, CC, 32, CC/16, abuf, 0, CC, nullptr, nullptr, 0, 0, 0);
  gemm(xv_h, CC, v1t_h, CC, 32, CC, 2,  nullptr, 0, 0, nullptr, t_h, 32, 32, 0);
  gemm(t_h, 32, v2t_h, 32, CC, 32, CC/16, vpre, 0, CC, nullptr, nullptr, 0, 0, 0);
  gemm(xg_h, CC, g1t_h, CC, 128, CC, 8, nullptr, 0, 0, nullptr, t_h, 128, 128, 2);   // sigmoid
  gemm(t_h, 128, g2t_h, 128, CC, 128, CC/16, gbuf, 0, CC, nullptr, nullptr, 0, 0, 0);

  pass2_kernel<<<BT*HH, 64, 0, stream>>>(kbuf, vbuf, abuf, wbuf, vpre, kkbuf,
                                         a0, w0, v0, k_k, k_a, vfirst);

  scan_kernel<<<BB*HH, 64, 0, stream>>>(rbuf, kbuf, vbuf, kkbuf, abuf, wbuf,
                                        state, ybuf, out + OFF_STATE);

  post_kernel<<<BT*HH, 64, 0, stream>>>(ybuf, rbuf, kbuf, vbuf, gbuf,
                                        ln_w, ln_b, r_k, gout_h);

  // final ghost linear into d_out
  gemm(gout_h, CC, outw_h, CC, CORE_P, CC, CORE_P/16, out, 0, CORE_N, out_b, core_h, CORE_P, CORE_N, 0);
  gemm(core_h, CORE_P, outg_h, CORE_P, GHOST_P, CORE_P, GHOST_P/16, out, CORE_N, GHOST_N, out_b, nullptr, 0, 0, 0);

  xlast_kernel<<<dim3((BB*CC+255)/256), 256, 0, stream>>>(x, out + OFF_XLAST);
  copyf_kernel<<<dim3((unsigned)((BTC+255)/256)), 256, 0, stream>>>(vfirst, out + OFF_VFIRST, BTC);
}
